// RNNEncoder_28948079575150
// MI455X (gfx1250) — matrix operand order repaired and checked by reading, not compiled, not run on hardware
//
#include <hip/hip_runtime.h>
#include <math.h>
#include <stdint.h>

#define USE_TDM 1

typedef __bf16 bhalf;
typedef __attribute__((ext_vector_type(16))) __bf16 v16bf;
typedef __attribute__((ext_vector_type(8)))  __bf16 v8bf;
typedef __attribute__((ext_vector_type(8)))  float  v8f;
typedef __attribute__((ext_vector_type(4)))  uint32_t u32x4;
typedef __attribute__((ext_vector_type(8)))  uint32_t u32x8;

#define T_SEQ 512
#define BATCH 64
#define DIN   512
#define HID   512
#define G4    2048
#define NTAG  50
#define M_ROWS (T_SEQ*BATCH)   /* 32768 */
#define LN_EPS 1e-5f

static __device__ __forceinline__ float sigmoidf_(float x){ return 1.f/(1.f+__expf(-x)); }

static __device__ __forceinline__ v8f wmma_bf16(v16bf a, v16bf b, v8f c){
  // v_wmma_f32_16x16x32_bf16  (neg_a, A, neg_b, B, c_mod, C, reuse_a, reuse_b)
  return __builtin_amdgcn_wmma_f32_16x16x32_bf16(false, a, false, b, (short)0, c, false, false);
}

// A-fragment (16x32 K-slice) per ISA 7.12.2 16-bit A layout.
static __device__ __forceinline__ v16bf load_a(const bhalf* base, int ld, int m0, int k0, int lane){
  const int m = lane & 15, hh = lane >> 4;
  const bhalf* p = base + (size_t)(m0 + m) * ld + k0 + hh * 8;
  v8bf lo = *(const v8bf*)p;
  v8bf hi = *(const v8bf*)(p + 16);
  return __builtin_shufflevector(lo, hi, 0,1,2,3,4,5,6,7,8,9,10,11,12,13,14,15);
}

// B-fragment (32x16 tile of B = W^T, W stored (N,K) row-major).
static __device__ __forceinline__ v16bf load_b(const bhalf* base, int ld, int n0, int k0, int lane){
  const int n = lane & 15, hh = lane >> 4;
  const bhalf* p = base + (size_t)(n0 + n) * ld + k0 + hh * 8;
  v8bf lo = *(const v8bf*)p;
  v8bf hi = *(const v8bf*)(p + 16);
  return __builtin_shufflevector(lo, hi, 0,1,2,3,4,5,6,7,8,9,10,11,12,13,14,15);
}

#if USE_TDM
// Issue TENSOR_LOAD_TO_LDS of a contiguous/2D bf16 tile. Must be executed by one wave.
// D# group0: [0]=count=1, [1]=lds byte addr, [2]=ga[31:0], [3]=ga[56:32]|type2.
// D# group1 bitfields per ISA 8.4 (data_size=2B).
static __device__ __forceinline__ void tdm_load_2d(uint64_t ga, uint32_t lds_addr,
        uint32_t tdim0, uint32_t tdim1, uint32_t tile0, uint32_t tile1, uint32_t stride0){
  u32x4 g0; u32x8 g1;
  g0[0] = 1u;
  g0[1] = lds_addr;
  g0[2] = (uint32_t)ga;
  g0[3] = (uint32_t)((ga >> 32) & 0x01ffffffu) | 0x80000000u;   // type=2 ("image")
  g1[0] = 0x00010000u;                                          // data_size=1 -> 2 bytes
  g1[1] = (tdim0 & 0xffffu) << 16;                              // tensor_dim0[15:0]
  g1[2] = ((tdim0 >> 16) & 0xffffu) | ((tdim1 & 0xffffu) << 16);// tensor_dim0[31:16], tensor_dim1[15:0]
  g1[3] = ((tdim1 >> 16) & 0xffffu) | ((tile0 & 0xffffu) << 16);// tensor_dim1[31:16], tile_dim0
  g1[4] = tile1 & 0xffffu;                                      // tile_dim1 (tile_dim2=0)
  g1[5] = stride0;                                              // tensor_dim0_stride[31:0]
  g1[6] = 0u; g1[7] = 0u;                                       // strides hi / dim1_stride
  asm volatile("tensor_load_to_lds %0, %1" :: "s"(g0), "s"(g1) : "memory");
}
#endif

// ---------------- conversion kernels ----------------

__global__ void k_xpose_cvt(const float* __restrict__ x, bhalf* __restrict__ xs){
  size_t gid = (size_t)blockIdx.x * blockDim.x + threadIdx.x;
  int i = (int)(gid % DIN);
  int b = (int)((gid / DIN) % BATCH);
  int t = (int)(gid / ((size_t)DIN * BATCH));
  xs[gid] = (bhalf)x[((size_t)b * T_SEQ + t) * DIN + i];
}

__global__ void k_cvt(const float* __restrict__ s, bhalf* __restrict__ d, int n){
  int gid = blockIdx.x * blockDim.x + threadIdx.x;
  if (gid < n) d[gid] = (bhalf)s[gid];
}

// ---------------- zih = LN(X @ W^T) ----------------
// one block = 32 rows of M, all 2048 columns; 16 waves = 2 Mx8 N-groups, 16 WMMA tiles each.
__global__ void __launch_bounds__(512) k_gemm_ln(const bhalf* __restrict__ X,
        const bhalf* __restrict__ W, const float* __restrict__ gam,
        const float* __restrict__ bet, bhalf* __restrict__ Z, int K){
  extern __shared__ char sm[];
  bhalf* As  = (bhalf*)sm;                                    // 32*K bf16 at LDS offset 0
  float* red = (float*)(sm + (size_t)32 * K * sizeof(bhalf)); // sum[32], sq[32]
  const int tid = threadIdx.x, lane = tid & 31, wv = tid >> 5;
  const int mt = wv & 1, ns = wv >> 1;
  const int m0g = blockIdx.x * 32;
  const int m0 = mt * 16;

  if (tid < 64) red[tid] = 0.f;
#if USE_TDM
  if (tid < 32){   // wave 0 issues the DMA stage of the A panel (contiguous 32*K elems)
    const uint32_t ne = 32u * (uint32_t)K;
    tdm_load_2d((uint64_t)(uintptr_t)(X + (size_t)m0g * K), 0u,
                ne, 1u, ne, 0u, ne);
    __builtin_amdgcn_s_wait_tensorcnt(0);
  }
#else
  {
    const int chunks = 32 * K / 8;
    const v8bf* src = (const v8bf*)(X + (size_t)m0g * K);
    v8bf* dst = (v8bf*)As;
    for (int c = tid; c < chunks; c += 512) dst[c] = src[c];
  }
#endif
  __syncthreads();

  v8f acc[16];
  v8f zz = {0.f,0.f,0.f,0.f,0.f,0.f,0.f,0.f};
  #pragma unroll
  for (int i = 0; i < 16; ++i) acc[i] = zz;

  for (int k0 = 0; k0 < K; k0 += 32){
    if (k0 + 32 < K)
      __builtin_prefetch(W + (size_t)(ns * 16 + (lane & 15)) * K + k0 + 32, 0, 3);
    v16bf a = load_a(As, K, m0, k0, lane);
    #pragma unroll
    for (int i = 0; i < 16; ++i){
      int n0 = (ns + 8 * i) * 16;
      v16bf b = load_b(W, K, n0, k0, lane);
      acc[i] = wmma_bf16(a, b, acc[i]);
    }
  }

  const int hh = lane >> 4, nn = lane & 15;
  #pragma unroll
  for (int r = 0; r < 8; ++r){
    int row = m0 + r + 8 * hh;
    float ps = 0.f, pq = 0.f;
    #pragma unroll
    for (int i = 0; i < 16; ++i){ float v = acc[i][r]; ps += v; pq += v * v; }
    atomicAdd(&red[row], ps);
    atomicAdd(&red[32 + row], pq);
  }
  __syncthreads();

  float mean[8], rst[8];
  #pragma unroll
  for (int r = 0; r < 8; ++r){
    int row = m0 + r + 8 * hh;
    float mu = red[row] * (1.f / 2048.f);
    float va = red[32 + row] * (1.f / 2048.f) - mu * mu;
    mean[r] = mu; rst[r] = rsqrtf(va + LN_EPS);
  }
  #pragma unroll
  for (int i = 0; i < 16; ++i){
    int col = (ns + 8 * i) * 16 + nn;
    float gm = gam[col], bt = bet[col];
    #pragma unroll
    for (int r = 0; r < 8; ++r){
      int m = m0g + m0 + r + 8 * hh;
      float val = gm * (acc[i][r] - mean[r]) * rst[r] + bt;
      Z[(size_t)m * G4 + col] = (bhalf)val;
    }
  }
}

// ---------------- persistent LSTM scan: one workgroup per direction ----------------
// LDS: gates 64x2048 bf16 (256KB, also TDM landing zone for zih[t]),
//      hbuf 64x512 bf16 (64KB; its first 1KB doubles as LN scratch while h is dead).
__global__ void __launch_bounds__(1024) k_scan(const bhalf* __restrict__ Zall,
        const bhalf* __restrict__ Whh, const float* __restrict__ lng,
        const float* __restrict__ lnb, const float* __restrict__ lncg,
        const float* __restrict__ lncb, bhalf* __restrict__ hout){
  const int d = blockIdx.x;
  const bhalf* Z  = Zall + (size_t)d * M_ROWS * G4;
  const bhalf* Wd = Whh  + (size_t)d * G4 * HID;
  const float* ghh = lng + ((size_t)d * 2 + 1) * G4;
  const float* bhh = lnb + ((size_t)d * 2 + 1) * G4;
  const float* gc  = lncg + (size_t)d * HID;
  const float* bc  = lncb + (size_t)d * HID;

  extern __shared__ char sm[];
  bhalf* gates = (bhalf*)sm;                                   // 64*2048 bf16 (LDS offset 0)
  bhalf* hbuf  = (bhalf*)(sm + (size_t)BATCH * G4 * 2);        // 64*512 bf16
  float* redR  = (float*)(sm + (size_t)BATCH * G4 * 2);        // 128 floats (aliases h row 0)
  float* redC  = redR + 128;                                   // 128 floats (aliases h row 0)

  const int tid = threadIdx.x, lane = tid & 31, wv = tid >> 5;
  const int mt = wv & 3, ns = wv >> 2;
  const int m0 = mt * 16;
  const int hh = lane >> 4, nn = lane & 15;

  float cst[32], og[32], hreg[32];
  #pragma unroll
  for (int r = 0; r < 32; ++r) cst[r] = 0.f;
  for (int e = tid; e < BATCH * HID; e += 1024) hbuf[e] = (bhalf)0.f;
  __syncthreads();

  for (int step = 0; step < T_SEQ; ++step){
    const int t = d ? (T_SEQ - 1 - step) : step;

#if USE_TDM
    // Overlap: DMA zih[t] (64x2048 bf16, contiguous) into gates while WMMA runs on hbuf.
    if (tid < 32)
      tdm_load_2d((uint64_t)(uintptr_t)(Z + (size_t)t * BATCH * G4), 0u,
                  (uint32_t)G4, (uint32_t)BATCH, (uint32_t)G4, (uint32_t)BATCH, (uint32_t)G4);
#endif

    // R = h_prev @ Whh^T   (64 x 2048), bf16 WMMA, Whh streamed from L2
    v8f acc[16];
    v8f zz = {0.f,0.f,0.f,0.f,0.f,0.f,0.f,0.f};
    #pragma unroll
    for (int i = 0; i < 16; ++i) acc[i] = zz;
    for (int k0 = 0; k0 < HID; k0 += 32){
      if (k0 + 32 < HID)
        __builtin_prefetch(Wd + (size_t)(ns * 16 + nn) * HID + k0 + 32, 0, 3);
      v16bf a = load_a(hbuf, HID, m0, k0, lane);
      #pragma unroll
      for (int i = 0; i < 16; ++i){
        int n0 = (ns + 8 * i) * 16;
        v16bf b = load_b(Wd, HID, n0, k0, lane);
        acc[i] = wmma_bf16(a, b, acc[i]);
      }
    }
    __syncthreads();                       // hbuf reads done -> scratch (h row 0) reusable
    if (tid < 128) redR[tid] = 0.f;
    __syncthreads();

    // LayerNorm over 2048 gate columns per batch row
    #pragma unroll
    for (int r = 0; r < 8; ++r){
      int row = m0 + r + 8 * hh;
      float ps = 0.f, pq = 0.f;
      #pragma unroll
      for (int i = 0; i < 16; ++i){ float v = acc[i][r]; ps += v; pq += v * v; }
      atomicAdd(&redR[row], ps);
      atomicAdd(&redR[64 + row], pq);
    }
    __syncthreads();
    float mean[8], rst[8];
    #pragma unroll
    for (int r = 0; r < 8; ++r){
      int row = m0 + r + 8 * hh;
      float mu = redR[row] * (1.f / 2048.f);
      float va = redR[64 + row] * (1.f / 2048.f) - mu * mu;
      mean[r] = mu; rst[r] = rsqrtf(va + LN_EPS);
    }
#if USE_TDM
    if (tid < 32) __builtin_amdgcn_s_wait_tensorcnt(0);
#endif
    __syncthreads();                       // zih[t] landed in gates; redR consumed

    // gates = zih[t] + LN(R)   (read DMA'd Z from LDS, overwrite in place)
    #pragma unroll
    for (int i = 0; i < 16; ++i){
      int col = (ns + 8 * i) * 16 + nn;
      float gm = ghh[col], bt = bhh[col];
      #pragma unroll
      for (int r = 0; r < 8; ++r){
        int bi = m0 + r + 8 * hh;
#if USE_TDM
        float zv = (float)gates[bi * G4 + col];
#else
        float zv = (float)Z[((size_t)t * BATCH + bi) * G4 + col];
#endif
        float val = gm * (acc[i][r] - mean[r]) * rst[r] + bt + zv;
        gates[bi * G4 + col] = (bhalf)val;
      }
    }
    __syncthreads();
    if (tid < 128) redC[tid] = 0.f;
    __syncthreads();

    // LSTM cell: c kept in registers; LN(c) reduction per batch row
    #pragma unroll
    for (int r = 0; r < 32; ++r){
      int idx = r * 1024 + tid;
      int bi = idx >> 9, j = idx & 511;
      float gi = (float)gates[bi * G4 + j];
      float gf = (float)gates[bi * G4 + 512 + j];
      float gg = (float)gates[bi * G4 + 1024 + j];
      float go = (float)gates[bi * G4 + 1536 + j];
      float cn = sigmoidf_(gf) * cst[r] + sigmoidf_(gi) * tanhf(gg);
      cst[r] = cn; og[r] = sigmoidf_(go);
      atomicAdd(&redC[bi], cn);
      atomicAdd(&redC[64 + bi], cn * cn);
    }
    __syncthreads();
    #pragma unroll
    for (int r = 0; r < 32; ++r){
      int idx = r * 1024 + tid;
      int bi = idx >> 9, j = idx & 511;
      float mu = redC[bi] * (1.f / 512.f);
      float va = redC[64 + bi] * (1.f / 512.f) - mu * mu;
      float rs = rsqrtf(va + LN_EPS);
      hreg[r] = og[r] * tanhf(gc[j] * (cst[r] - mu) * rs + bc[j]);
    }
    __syncthreads();                       // redC consumed; hbuf writable
    #pragma unroll
    for (int r = 0; r < 32; ++r){
      int idx = r * 1024 + tid;
      int bi = idx >> 9, j = idx & 511;
      bhalf hv = (bhalf)hreg[r];
      hbuf[bi * HID + j] = hv;
      hout[((size_t)t * BATCH + bi) * (2 * HID) + d * HID + j] = hv;
    }
    __syncthreads();
  }
}

// ---------------- attention + tag softmax ----------------

__global__ void __launch_bounds__(256) k_att_scores(const bhalf* __restrict__ mb,
        const float* __restrict__ aw, float* __restrict__ a){
  __shared__ float awsh[512];
  __shared__ float red[256];
  const int b = blockIdx.x, tid = threadIdx.x;
  for (int h = tid; h < 512; h += 256) awsh[h] = aw[(size_t)b * 512 + h];
  __syncthreads();
  float loc[4];
  #pragma unroll
  for (int q = 0; q < 4; ++q){
    int l = tid + 256 * q;
    float s = 0.f;
    const bhalf* col = mb + (size_t)b * 512 * 1024 + l;
    for (int h = 0; h < 512; ++h) s += awsh[h] * tanhf((float)col[(size_t)h * 1024]);
    loc[q] = s;
  }
  float mx = fmaxf(fmaxf(loc[0], loc[1]), fmaxf(loc[2], loc[3]));
  red[tid] = mx; __syncthreads();
  for (int s = 128; s > 0; s >>= 1){ if (tid < s) red[tid] = fmaxf(red[tid], red[tid + s]); __syncthreads(); }
  mx = red[0]; __syncthreads();
  float sum = 0.f;
  #pragma unroll
  for (int q = 0; q < 4; ++q){ loc[q] = __expf(loc[q] - mx); sum += loc[q]; }
  red[tid] = sum; __syncthreads();
  for (int s = 128; s > 0; s >>= 1){ if (tid < s) red[tid] += red[tid + s]; __syncthreads(); }
  float inv = 1.f / red[0];
  #pragma unroll
  for (int q = 0; q < 4; ++q) a[(size_t)b * 1024 + tid + 256 * q] = loc[q] * inv;
}

__global__ void __launch_bounds__(512) k_att_out(const bhalf* __restrict__ mb,
        const float* __restrict__ a, float* __restrict__ ao){
  __shared__ float ash[1024];
  const int b = blockIdx.x, tid = threadIdx.x;
  for (int l = tid; l < 1024; l += 512) ash[l] = a[(size_t)b * 1024 + l];
  __syncthreads();
  const bhalf* row = mb + ((size_t)b * 512 + tid) * 1024;
  float s = 0.f;
  for (int l = 0; l < 1024; ++l) s += (float)row[l] * ash[l];
  ao[(size_t)b * 512 + tid] = tanhf(s);
}

__global__ void __launch_bounds__(64) k_final(const float* __restrict__ rel,
        const float* __restrict__ rbias, const float* __restrict__ ao,
        float* __restrict__ out){
  __shared__ float sv[64];
  __shared__ float mxs, sms;
  const int b = blockIdx.x, tid = threadIdx.x;
  float s = -1e30f;
  if (tid < NTAG){
    float accv = 0.f;
    const float* rrow = rel + (size_t)tid * 512;
    const float* arow = ao + (size_t)b * 512;
    for (int h = 0; h < 512; ++h) accv += rrow[h] * arow[h];
    s = accv + rbias[(size_t)b * NTAG + tid];
  }
  sv[tid] = s; __syncthreads();
  if (tid == 0){
    float m = -1e30f; for (int i = 0; i < NTAG; ++i) m = fmaxf(m, sv[i]);
    float su = 0.f;   for (int i = 0; i < NTAG; ++i) su += __expf(sv[i] - m);
    mxs = m; sms = su;
  }
  __syncthreads();
  if (tid < NTAG) out[(size_t)b * NTAG + tid] = __expf(s - mxs) / sms;
}

// ---------------- launcher ----------------

extern "C" void kernel_launch(void* const* d_in, const int* in_sizes, int n_in,
                              void* d_out, int out_size, void* d_ws, size_t ws_size,
                              hipStream_t stream){
  (void)in_sizes; (void)n_in; (void)out_size; (void)ws_size;
  const float* x     = (const float*)d_in[0];
  const float* Wih0  = (const float*)d_in[2];
  const float* Whh0  = (const float*)d_in[3];
  const float* lng0  = (const float*)d_in[4];
  const float* lnb0  = (const float*)d_in[5];
  const float* lncg0 = (const float*)d_in[6];
  const float* lncb0 = (const float*)d_in[7];
  const float* Wih1  = (const float*)d_in[8];
  const float* Whh1  = (const float*)d_in[9];
  const float* lng1  = (const float*)d_in[10];
  const float* lnb1  = (const float*)d_in[11];
  const float* lncg1 = (const float*)d_in[12];
  const float* lncb1 = (const float*)d_in[13];
  const float* attw  = (const float*)d_in[14];
  const float* rel   = (const float*)d_in[15];
  const float* rbias = (const float*)d_in[16];
  float* out = (float*)d_out;

  char* ws = (char*)d_ws;
  size_t off = 0;
  auto take = [&](size_t bytes)->char* {
    char* p = ws + off; off = (off + bytes + 255) & ~(size_t)255; return p;
  };
  bhalf* xs   = (bhalf*)take((size_t)M_ROWS * DIN * 2);
  bhalf* wih0 = (bhalf*)take((size_t)2 * G4 * DIN * 2);
  bhalf* whh0 = (bhalf*)take((size_t)2 * G4 * HID * 2);
  bhalf* wih1 = (bhalf*)take((size_t)2 * G4 * (2 * HID) * 2);
  bhalf* whh1 = (bhalf*)take((size_t)2 * G4 * HID * 2);
  bhalf* Zbuf = (bhalf*)take((size_t)2 * M_ROWS * G4 * 2);
  bhalf* h1   = (bhalf*)take((size_t)M_ROWS * (2 * HID) * 2);
  bhalf* mbuf = (bhalf*)take((size_t)M_ROWS * (2 * HID) * 2);
  float* asc  = (float*)take((size_t)BATCH * 1024 * 4);
  float* ao   = (float*)take((size_t)BATCH * HID * 4);

  k_xpose_cvt<<<(M_ROWS * DIN) / 256, 256, 0, stream>>>(x, xs);
  k_cvt<<<(2 * G4 * DIN) / 256, 256, 0, stream>>>(Wih0, wih0, 2 * G4 * DIN);
  k_cvt<<<(2 * G4 * HID) / 256, 256, 0, stream>>>(Whh0, whh0, 2 * G4 * HID);
  k_cvt<<<(2 * G4 * 2 * HID) / 256, 256, 0, stream>>>(Wih1, wih1, 2 * G4 * 2 * HID);
  k_cvt<<<(2 * G4 * HID) / 256, 256, 0, stream>>>(Whh1, whh1, 2 * G4 * HID);

  const size_t scan_smem = (size_t)BATCH * G4 * 2 + (size_t)BATCH * HID * 2; // 320KB
  for (int dd = 0; dd < 2; ++dd)
    k_gemm_ln<<<M_ROWS / 32, 512, 32 * DIN * 2 + 256, stream>>>(
        xs, wih0 + (size_t)dd * G4 * DIN,
        lng0 + (size_t)dd * 2 * G4, lnb0 + (size_t)dd * 2 * G4,
        Zbuf + (size_t)dd * M_ROWS * G4, DIN);
  k_scan<<<2, 1024, scan_smem, stream>>>(Zbuf, whh0, lng0, lnb0, lncg0, lncb0, h1);

  for (int dd = 0; dd < 2; ++dd)
    k_gemm_ln<<<M_ROWS / 32, 512, 32 * (2 * HID) * 2 + 256, stream>>>(
        h1, wih1 + (size_t)dd * G4 * (2 * HID),
        lng1 + (size_t)dd * 2 * G4, lnb1 + (size_t)dd * 2 * G4,
        Zbuf + (size_t)dd * M_ROWS * G4, 2 * HID);
  k_scan<<<2, 1024, scan_smem, stream>>>(Zbuf, whh1, lng1, lnb1, lncg1, lncb1, mbuf);

  k_att_scores<<<BATCH, 256, 0, stream>>>(mbuf, attw, asc);
  k_att_out<<<BATCH, 512, 0, stream>>>(mbuf, asc, ao);
  k_final<<<BATCH, 64, 0, stream>>>(rel, rbias, ao, out);
}
